// MetaSystem_16793322127886
// MI455X (gfx1250) — compile-verified, matrix operand order checked by reading, not run on hardware
//
#include <hip/hip_runtime.h>

// ---------------------------------------------------------------------------
// MI455X (gfx1250) implementation. Memory-bound workload (~1.3 GB traffic at
// 23.3 TB/s ~ 56us); GEMMs (155 GFLOP) done in bf16 on the WMMA pipes so the
// matrix math is never the bottleneck. wave32, 8 waves / 256-thread block.
// Round 2: LDS double-buffered A tiles (1 barrier per K-step) + preloaded B
// fragments so the 4 WMMAs per step issue back-to-back after a single wait.
// ---------------------------------------------------------------------------

typedef __attribute__((ext_vector_type(16))) __bf16          v16bf;
typedef __attribute__((ext_vector_type(8)))  float           v8f;
typedef __attribute__((ext_vector_type(8)))  unsigned short  ushort8;

#define B_SZ 32768
#define H_SZ 1024
#define D_SZ 512
#define M_SZ 8
#define K_G  (H_SZ + D_SZ)   // 1536, gate GEMM K
#define LN_EPS 1e-5f
#define LDA 40               // LDS A-tile row stride (ushorts): 16B-aligned, bank-skewed

__device__ __forceinline__ unsigned short f2bf(float f) {
  unsigned int u = __builtin_bit_cast(unsigned int, f);
  u += 0x7FFFu + ((u >> 16) & 1u);            // round-to-nearest-even
  return (unsigned short)(u >> 16);
}

// Assemble a 16x32 bf16 WMMA fragment (A or B operand) from two contiguous
// 8-element (16-byte) runs, per the CDNA5 16-bit operand VGPR layout.
union BFrag {
  struct { ushort8 lo, hi; } p;
  v16bf v;
};

__device__ __forceinline__ v16bf frag_from(const unsigned short* row, int h8) {
  BFrag f;
  f.p.lo = *(const ushort8*)(row + h8);        // K = h8 .. h8+7
  f.p.hi = *(const ushort8*)(row + 16 + h8);   // K = 16+h8 .. 16+h8+7
  return f.v;
}

__device__ __forceinline__ v8f vzero8() {
  v8f z;
  #pragma unroll
  for (int i = 0; i < 8; ++i) z[i] = 0.0f;
  return z;
}

__device__ __forceinline__ float sigmoidf_(float x) {
  return 1.0f / (1.0f + __expf(-x));
}

// ---------------------------------------------------------------------------
// K0: convert weights to transposed bf16 (WT[n*K + k] = bf16(W[k*N + n])).
// Small (<3MB) -> stays hot in the 192MB L2 for all GEMM blocks.
// ---------------------------------------------------------------------------
__global__ void k_prep(const float* __restrict__ Wp, const float* __restrict__ Wg,
                       const float* __restrict__ Wc1,
                       unsigned short* __restrict__ WpT,
                       unsigned short* __restrict__ WgT,
                       unsigned short* __restrict__ Wc1T) {
  long i = (long)blockIdx.x * blockDim.x + threadIdx.x;
  const long nWp = (long)D_SZ * H_SZ;          // 512 x 1024
  const long nWg = (long)D_SZ * K_G;           // 512 x 1536
  const long nWc = 256L * D_SZ;                // 256 x 512
  if (i < nWp) {
    int n = (int)(i / H_SZ), k = (int)(i % H_SZ);
    WpT[i] = f2bf(Wp[(long)k * D_SZ + n]);
  } else if (i < nWp + nWg) {
    long j = i - nWp;
    int n = (int)(j / K_G), k = (int)(j % K_G);
    WgT[j] = f2bf(Wg[(long)k * D_SZ + n]);
  } else if (i < nWp + nWg + nWc) {
    long j = i - nWp - nWg;
    int n = (int)(j / D_SZ), k = (int)(j % D_SZ);
    Wc1T[j] = f2bf(Wc1[(long)k * 256 + n]);
  }
}

// ---------------------------------------------------------------------------
// K1: h = hidden @ Wp + bp  (WMMA bf16), fused LayerNorm -> new_h (f32, ws).
// Block = 16 rows x 512 cols; wave w owns cols [w*64, w*64+64) = 4 WMMA tiles.
// Double-buffered A staging: one barrier per K-step, next tile's global
// loads overlap the current WMMA chain.
// ---------------------------------------------------------------------------
__global__ void __launch_bounds__(256) k_proj_ln(
    const float* __restrict__ hidden, const unsigned short* __restrict__ WpT,
    const float* __restrict__ bp, const float* __restrict__ ln_g,
    const float* __restrict__ ln_b, float* __restrict__ newh) {
  __shared__ unsigned short aT[2][16 * LDA];
  __shared__ float hT[16 * D_SZ];
  __shared__ float pSum[16][16], pSq[16][16];
  __shared__ float muS[16], rsS[16];

  const int t = threadIdx.x, w = t >> 5, l = t & 31;
  const int r0 = blockIdx.x * 16;
  const int m_l = l & 15, h8 = (l >> 4) * 8;
  // this thread's 2 staged elements: e = 2t, 2t+1
  const int sm0 = (t * 2) >> 5, sk0 = (t * 2) & 31;
  const int sm1 = (t * 2 + 1) >> 5, sk1 = (t * 2 + 1) & 31;
  const float* srow0 = hidden + (long)(r0 + sm0) * H_SZ;
  const float* srow1 = hidden + (long)(r0 + sm1) * H_SZ;

  v8f acc[4];
  #pragma unroll
  for (int i = 0; i < 4; ++i) acc[i] = vzero8();

  // prologue: stage K-step 0 into buffer 0
  aT[0][sm0 * LDA + sk0] = f2bf(srow0[sk0]);
  aT[0][sm1 * LDA + sk1] = f2bf(srow1[sk1]);

  int cur = 0;
  for (int k0 = 0; k0 < H_SZ; k0 += 32) {
    const bool more = (k0 + 32) < H_SZ;
    float nx0 = 0.f, nx1 = 0.f;
    if (more) {                       // issue next tile's global loads early
      nx0 = srow0[k0 + 32 + sk0];
      nx1 = srow1[k0 + 32 + sk1];
    }
    __syncthreads();                  // aT[cur] ready for all waves
    v16bf a = frag_from(&aT[cur][m_l * LDA], h8);
    v16bf bf[4];
    #pragma unroll
    for (int nt = 0; nt < 4; ++nt) {  // preload all B frags (one load clause)
      int n = w * 64 + nt * 16 + m_l;
      bf[nt] = frag_from(&WpT[(long)n * H_SZ + k0], h8);
    }
    #pragma unroll
    for (int nt = 0; nt < 4; ++nt)    // back-to-back WMMAs
      acc[nt] = __builtin_amdgcn_wmma_f32_16x16x32_bf16(
          false, a, false, bf[nt], (short)0, acc[nt], false, false);
    if (more) {                       // write next tile into the other buffer
      aT[cur ^ 1][sm0 * LDA + sk0] = f2bf(nx0);
      aT[cur ^ 1][sm1 * LDA + sk1] = f2bf(nx1);
    }
    cur ^= 1;
  }

  // spill tile (+bias) to LDS for the row-wise LayerNorm
  __syncthreads();
  #pragma unroll
  for (int nt = 0; nt < 4; ++nt) {
    int n = w * 64 + nt * 16 + m_l;
    float bpn = bp[n];
    #pragma unroll
    for (int v = 0; v < 8; ++v) {
      int m = (l >> 4) * 8 + v;
      hT[m * D_SZ + n] = acc[nt][v] + bpn;
    }
  }
  __syncthreads();

  {   // 16 threads per row: partial sum / sumsq over 32 elements each
    int row = t >> 4, part = t & 15;
    float s = 0.f, sq = 0.f;
    #pragma unroll
    for (int i = 0; i < 32; ++i) {
      float x = hT[row * D_SZ + part * 32 + i];
      s += x; sq += x * x;
    }
    pSum[row][part] = s; pSq[row][part] = sq;
  }
  __syncthreads();
  if (t < 16) {
    float s = 0.f, sq = 0.f;
    #pragma unroll
    for (int i = 0; i < 16; ++i) { s += pSum[t][i]; sq += pSq[t][i]; }
    float mu = s * (1.0f / D_SZ);
    float var = sq * (1.0f / D_SZ) - mu * mu;
    muS[t] = mu; rsS[t] = rsqrtf(var + LN_EPS);
  }
  __syncthreads();
  {   // vectorized LN epilogue: 8 float4 stores per thread
    int row = t >> 4, part = t & 15;
    float mu = muS[row], rs = rsS[row];
    const float4* hT4  = (const float4*)&hT[row * D_SZ];
    const float4* g4   = (const float4*)ln_g;
    const float4* b4   = (const float4*)ln_b;
    float4* out4 = (float4*)(newh + (long)(r0 + row) * D_SZ);
    #pragma unroll
    for (int i = 0; i < 8; ++i) {
      int n4 = part * 8 + i;
      float4 x = hT4[n4], g = g4[n4], bb = b4[n4];
      float4 r;
      r.x = (x.x - mu) * rs * g.x + bb.x;
      r.y = (x.y - mu) * rs * g.y + bb.y;
      r.z = (x.z - mu) * rs * g.z + bb.z;
      r.w = (x.w - mu) * rs * g.w + bb.w;
      out4[n4] = r;
    }
  }
}

// ---------------------------------------------------------------------------
// K2: conf = sigmoid(relu(hyp @ Wc1 + bc1) @ Wc2 + bc2)  + argmin/argmax.
// hyp viewed as (B*M, 512); block = 16 flat rows (2 batches) x 256 cols.
// Wave w owns cols [w*32, w*32+32) = 2 WMMA tiles. ReLU*Wc2 reduced with
// ds_add_f32 atomics into per-row partials.
// ---------------------------------------------------------------------------
__global__ void __launch_bounds__(256) k_conf(
    const float* __restrict__ hyp, const unsigned short* __restrict__ Wc1T,
    const float* __restrict__ bc1, const float* __restrict__ Wc2,
    const float* __restrict__ bc2, float* __restrict__ conf_out,
    int* __restrict__ minIdx, int* __restrict__ maxIdx) {
  __shared__ unsigned short aT[2][16 * LDA];
  __shared__ float rowS[16];
  __shared__ float confS[16];

  const int t = threadIdx.x, w = t >> 5, l = t & 31;
  const long r0 = (long)blockIdx.x * 16;        // flat row in (B*M)
  const int m_l = l & 15, h8 = (l >> 4) * 8;
  const int sm0 = (t * 2) >> 5, sk0 = (t * 2) & 31;
  const int sm1 = (t * 2 + 1) >> 5, sk1 = (t * 2 + 1) & 31;
  const float* srow0 = hyp + (r0 + sm0) * D_SZ;
  const float* srow1 = hyp + (r0 + sm1) * D_SZ;
  if (t < 16) rowS[t] = 0.f;

  v8f acc[2];
  acc[0] = vzero8(); acc[1] = vzero8();

  aT[0][sm0 * LDA + sk0] = f2bf(srow0[sk0]);
  aT[0][sm1 * LDA + sk1] = f2bf(srow1[sk1]);

  int cur = 0;
  for (int k0 = 0; k0 < D_SZ; k0 += 32) {
    const bool more = (k0 + 32) < D_SZ;
    float nx0 = 0.f, nx1 = 0.f;
    if (more) {
      nx0 = srow0[k0 + 32 + sk0];
      nx1 = srow1[k0 + 32 + sk1];
    }
    __syncthreads();
    v16bf a = frag_from(&aT[cur][m_l * LDA], h8);
    v16bf bf[2];
    #pragma unroll
    for (int nt = 0; nt < 2; ++nt) {
      int n = w * 32 + nt * 16 + m_l;
      bf[nt] = frag_from(&Wc1T[(long)n * D_SZ + k0], h8);
    }
    #pragma unroll
    for (int nt = 0; nt < 2; ++nt)
      acc[nt] = __builtin_amdgcn_wmma_f32_16x16x32_bf16(
          false, a, false, bf[nt], (short)0, acc[nt], false, false);
    if (more) {
      aT[cur ^ 1][sm0 * LDA + sk0] = f2bf(nx0);
      aT[cur ^ 1][sm1 * LDA + sk1] = f2bf(nx1);
    }
    cur ^= 1;
  }

  __syncthreads();
  #pragma unroll
  for (int nt = 0; nt < 2; ++nt) {
    int n = w * 32 + nt * 16 + m_l;
    float b1 = bc1[n], w2 = Wc2[n];
    #pragma unroll
    for (int v = 0; v < 8; ++v) {
      int m = (l >> 4) * 8 + v;
      float x = acc[nt][v] + b1;
      x = x > 0.f ? x : 0.f;                   // ReLU
      atomicAdd(&rowS[m], x * w2);             // ds_add_f32
    }
  }
  __syncthreads();
  if (t < 16) {
    float c = sigmoidf_(rowS[t] + bc2[0]);
    conf_out[r0 + t] = c;
    confS[t] = c;
  }
  __syncthreads();
  if (t < 2) {                                  // 2 batches per block
    int base = t * 8;
    float mn = confS[base], mx = confS[base];
    int mni = 0, mxi = 0;
    #pragma unroll
    for (int j = 1; j < M_SZ; ++j) {
      float c = confS[base + j];
      if (c < mn) { mn = c; mni = j; }          // first occurrence, like argmin
      if (c > mx) { mx = c; mxi = j; }
    }
    long bidx = (r0 >> 3) + t;
    minIdx[bidx] = mni;
    maxIdx[bidx] = mxi;
  }
}

// ---------------------------------------------------------------------------
// K3: gate = sigmoid([hidden, new_h] @ Wg + bg); writes the blended
// replacement row of `updated` directly at slot min_idx[b].
// K loop spans 1536: A tiles spliced from hidden (k<1024) and new_h.
// ---------------------------------------------------------------------------
__global__ void __launch_bounds__(256) k_gate_update(
    const float* __restrict__ hidden, const float* __restrict__ newh,
    const unsigned short* __restrict__ WgT, const float* __restrict__ bg,
    const float* __restrict__ hyp, const int* __restrict__ minIdx,
    float* __restrict__ updated) {
  __shared__ unsigned short aT[2][16 * LDA];
  const int t = threadIdx.x, w = t >> 5, l = t & 31;
  const int r0 = blockIdx.x * 16;
  const int m_l = l & 15, h8 = (l >> 4) * 8;
  const int sm0 = (t * 2) >> 5, sk0 = (t * 2) & 31;
  const int sm1 = (t * 2 + 1) >> 5, sk1 = (t * 2 + 1) & 31;

  auto src = [&](int m, int kk) -> float {
    return (kk < H_SZ) ? hidden[(long)(r0 + m) * H_SZ + kk]
                       : newh[(long)(r0 + m) * D_SZ + (kk - H_SZ)];
  };

  v8f acc[4];
  #pragma unroll
  for (int i = 0; i < 4; ++i) acc[i] = vzero8();

  aT[0][sm0 * LDA + sk0] = f2bf(src(sm0, sk0));
  aT[0][sm1 * LDA + sk1] = f2bf(src(sm1, sk1));

  int cur = 0;
  for (int k0 = 0; k0 < K_G; k0 += 32) {
    const bool more = (k0 + 32) < K_G;
    float nx0 = 0.f, nx1 = 0.f;
    if (more) {
      nx0 = src(sm0, k0 + 32 + sk0);
      nx1 = src(sm1, k0 + 32 + sk1);
    }
    __syncthreads();
    v16bf a = frag_from(&aT[cur][m_l * LDA], h8);
    v16bf bf[4];
    #pragma unroll
    for (int nt = 0; nt < 4; ++nt) {
      int n = w * 64 + nt * 16 + m_l;
      bf[nt] = frag_from(&WgT[(long)n * K_G + k0], h8);
    }
    #pragma unroll
    for (int nt = 0; nt < 4; ++nt)
      acc[nt] = __builtin_amdgcn_wmma_f32_16x16x32_bf16(
          false, a, false, bf[nt], (short)0, acc[nt], false, false);
    if (more) {
      aT[cur ^ 1][sm0 * LDA + sk0] = f2bf(nx0);
      aT[cur ^ 1][sm1 * LDA + sk1] = f2bf(nx1);
    }
    cur ^= 1;
  }

  #pragma unroll
  for (int nt = 0; nt < 4; ++nt) {
    int n = w * 64 + nt * 16 + m_l;
    float bgn = bg[n];
    #pragma unroll
    for (int v = 0; v < 8; ++v) {
      int m = (l >> 4) * 8 + v;
      long b = r0 + m;
      float g  = sigmoidf_(acc[nt][v] + bgn);
      int  mi  = minIdx[b];
      long slot = (b * M_SZ + mi) * D_SZ + n;
      float old = hyp[slot];
      float nh  = newh[b * D_SZ + n];
      updated[slot] = g * old + (1.0f - g) * nh;
    }
  }
}

// ---------------------------------------------------------------------------
// K4: stream-copy the 7/8 untouched hypothesis rows into `updated` (float4).
// ---------------------------------------------------------------------------
__global__ void k_copy_updated(const float4* __restrict__ hyp4,
                               const int* __restrict__ minIdx,
                               float4* __restrict__ upd4) {
  long i = (long)blockIdx.x * blockDim.x + threadIdx.x;   // over B*M*D/4
  const int D4 = D_SZ / 4;
  long row = i / D4;                    // flat (B*M)
  int  j   = (int)(row & (M_SZ - 1));
  long b   = row >> 3;
  if (j != minIdx[b]) upd4[i] = hyp4[i];
}

// ---------------------------------------------------------------------------
// K5: primary[b,:] = updated[b, max_idx[b], :]  (float4 gather).
// ---------------------------------------------------------------------------
__global__ void k_primary(const float4* __restrict__ upd4,
                          const int* __restrict__ maxIdx,
                          float4* __restrict__ prim4) {
  long i = (long)blockIdx.x * blockDim.x + threadIdx.x;   // over B*D/4
  const int D4 = D_SZ / 4;
  long b  = i / D4;
  int  d4 = (int)(i % D4);
  int  mx = maxIdx[b];
  prim4[i] = upd4[(b * M_SZ + mx) * D4 + d4];
}

// ---------------------------------------------------------------------------
extern "C" void kernel_launch(void* const* d_in, const int* in_sizes, int n_in,
                              void* d_out, int out_size, void* d_ws, size_t ws_size,
                              hipStream_t stream) {
  (void)in_sizes; (void)n_in; (void)out_size; (void)ws_size;

  const float* hidden = (const float*)d_in[0];
  const float* hyp    = (const float*)d_in[1];
  const float* Wp     = (const float*)d_in[2];
  const float* bp     = (const float*)d_in[3];
  const float* ln_g   = (const float*)d_in[4];
  const float* ln_b   = (const float*)d_in[5];
  const float* Wc1    = (const float*)d_in[6];
  const float* bc1    = (const float*)d_in[7];
  const float* Wc2    = (const float*)d_in[8];
  const float* bc2    = (const float*)d_in[9];
  const float* Wg     = (const float*)d_in[10];
  const float* bg     = (const float*)d_in[11];

  float* updated = (float*)d_out;                          // (B, M, D)
  float* primary = updated + (long)B_SZ * M_SZ * D_SZ;     // (B, D)
  float* conf    = primary + (long)B_SZ * D_SZ;            // (B, M)

  // workspace carve-out (~70 MB), 256B aligned blocks
  char* ws = (char*)d_ws;
  size_t off = 0;
  auto carve = [&](size_t bytes) -> char* {
    off = (off + 255) & ~(size_t)255;
    char* p = ws + off;
    off += bytes;
    return p;
  };
  unsigned short* WpT  = (unsigned short*)carve(sizeof(unsigned short) * (size_t)D_SZ * H_SZ);
  unsigned short* WgT  = (unsigned short*)carve(sizeof(unsigned short) * (size_t)D_SZ * K_G);
  unsigned short* Wc1T = (unsigned short*)carve(sizeof(unsigned short) * 256 * (size_t)D_SZ);
  float*          newh = (float*)carve(sizeof(float) * (size_t)B_SZ * D_SZ);
  int*            mnI  = (int*)carve(sizeof(int) * B_SZ);
  int*            mxI  = (int*)carve(sizeof(int) * B_SZ);

  long nPrep = (long)D_SZ * H_SZ + (long)D_SZ * K_G + 256L * D_SZ;
  k_prep<<<(int)((nPrep + 255) / 256), 256, 0, stream>>>(Wp, Wg, Wc1, WpT, WgT, Wc1T);

  k_proj_ln<<<B_SZ / 16, 256, 0, stream>>>(hidden, WpT, bp, ln_g, ln_b, newh);

  k_conf<<<B_SZ * M_SZ / 16, 256, 0, stream>>>(hyp, Wc1T, bc1, Wc2, bc2, conf, mnI, mxI);

  k_gate_update<<<B_SZ / 16, 256, 0, stream>>>(hidden, newh, WgT, bg, hyp, mnI, updated);

  long n4 = (long)B_SZ * M_SZ * D_SZ / 4;
  k_copy_updated<<<(int)((n4 + 255) / 256), 256, 0, stream>>>(
      (const float4*)hyp, mnI, (float4*)updated);

  long p4 = (long)B_SZ * D_SZ / 4;
  k_primary<<<(int)((p4 + 255) / 256), 256, 0, stream>>>(
      (const float4*)updated, mxI, (float4*)primary);
}
